// graph_maker_41343355191810
// MI455X (gfx1250) — matrix operand order repaired and checked by reading, not compiled, not run on hardware
//
#include <hip/hip_runtime.h>
#include <hip/hip_bf16.h>
#include <math.h>
#include <stdint.h>

#define N_USERS 6144
#define M_ITEMS 6144
#define D_FEAT  512
#define NNZ     500000
#define TOPK    20
#define N_NODES (N_USERS + M_ITEMS)
#define TILE    16
#define CTILES  4                    // column tiles per wave pass
#define CBLK    (TILE * CTILES)      // 64 columns per pass
#define NCGRP   (M_ITEMS / CBLK)     // 96 passes

typedef __attribute__((ext_vector_type(16))) _Float16 v16h;
typedef __attribute__((ext_vector_type(8)))  _Float16 v8h;
typedef __attribute__((ext_vector_type(8)))  float    v8f;
typedef __attribute__((ext_vector_type(4)))  unsigned int v4u;
typedef __attribute__((ext_vector_type(8)))  int      v8i;
typedef __attribute__((ext_vector_type(4)))  int      v4i;

#if defined(__has_builtin)
#if __has_builtin(__builtin_amdgcn_tensor_load_to_lds)
#define HAVE_TDM 1
#endif
#endif
#if __has_include(<hip/amd_detail/amd_gfx1250_TDM.h>)
#define TDM_6ARG 1   // amdgpu-toolchain (clang-23 / therock headers)
#endif

// ---------------------------------------------------------------------------
// Kernel 1: h = relu(x*w0)*w1 ; emb = h / max(||h||, 1e-12)  -> f16 into ws
// ---------------------------------------------------------------------------
__global__ __launch_bounds__(256) void embed_normalize_kernel(
    const float* __restrict__ x, const float* __restrict__ w0,
    const float* __restrict__ w1, _Float16* __restrict__ emb) {
  __shared__ float red[256];
  const int row = blockIdx.x;
  const int t = threadIdx.x;
  const float* xr = x + (size_t)row * D_FEAT;

  float v0 = xr[t] * w0[t];
  v0 = v0 > 0.0f ? v0 : 0.0f;
  v0 *= w1[t];
  float v1 = xr[t + 256] * w0[t + 256];
  v1 = v1 > 0.0f ? v1 : 0.0f;
  v1 *= w1[t + 256];

  red[t] = v0 * v0 + v1 * v1;
  __syncthreads();
  for (int s = 128; s > 0; s >>= 1) {
    if (t < s) red[t] += red[t + s];
    __syncthreads();
  }
  const float scale = 1.0f / fmaxf(sqrtf(red[0]), 1e-12f);
  emb[(size_t)row * D_FEAT + t]       = (_Float16)(v0 * scale);
  emb[(size_t)row * D_FEAT + t + 256] = (_Float16)(v1 * scale);
}

// ---------------------------------------------------------------------------
// Kernel 2: scatter base graph (values replaced by ones, duplicates sum)
// ---------------------------------------------------------------------------
__global__ __launch_bounds__(256) void edge_scatter_kernel(
    const long long* __restrict__ gi, float* __restrict__ adj) {
  const int e = blockIdx.x * blockDim.x + threadIdx.x;
  if (e >= NNZ) return;
  const long long r = gi[e];        // user row  [0, N_USERS)
  const long long c = gi[NNZ + e];  // item col  [N_USERS, N_NODES)
  atomicAdd(adj + (size_t)r * N_NODES + (size_t)c, 1.0f);
}

// ---------------------------------------------------------------------------
// Kernel 3: fused  sim = emb @ emb^T  (WMMA f16->f32, 4 accumulators/wave)
//           ->  per-row top-20 -> thresholded symmetric scatter.
// One wave32 per 16-row block; A staged via TDM tensor_load_to_lds.
// ---------------------------------------------------------------------------
__global__ __launch_bounds__(32) void simtopk_scatter_kernel(
    const _Float16* __restrict__ emb, const float* __restrict__ bptr,
    float* __restrict__ adj) {
  __shared__ _Float16 As[TILE * D_FEAT];   // 16 KB: 16 rows of A in f16
  __shared__ float tilebuf[TILE][CBLK];    // one 16x64 sim strip (4 KB)
  __shared__ float topv[TILE][TOPK];
  __shared__ int   topi[TILE][TOPK];

  const int lane = threadIdx.x;
  const int row0 = blockIdx.x * TILE;

#if defined(HAVE_TDM)
  // --- Tensor Data Mover: DMA the 16x512 f16 A block into LDS -------------
  {
    const unsigned lds_off =
        (unsigned)(uintptr_t)(&As[0]);  // low 32 bits of LDS aperture addr
    const unsigned long long ga =
        (unsigned long long)(uintptr_t)(emb + (size_t)row0 * D_FEAT);
    // D# group 0: count=1 | lds_addr | global_addr[56:0] | type=2
    v4u g0 = { 1u, lds_off, (unsigned)(ga & 0xffffffffu),
               (unsigned)((ga >> 32) & 0x1ffffffu) | 0x80000000u };
    // D# group 1: data_size=2B; tensor_dim0=512, tensor_dim1=16;
    //             tile_dim0=512, tile_dim1=16; dim0_stride=512
    v8i g1 = { 0x00010000, (int)(512u << 16), (int)(16u << 16),
               (int)(512u << 16), 16, 512, 0, 0 };
    v4i z4 = { 0, 0, 0, 0 };
#if defined(TDM_6ARG)
    v8i z8 = { 0, 0, 0, 0, 0, 0, 0, 0 };
    __builtin_amdgcn_tensor_load_to_lds(g0, g1, z4, z4, z8, 0);
#else
    __builtin_amdgcn_tensor_load_to_lds(g0, g1, z4, z4, 0);
#endif
  }
#else
  {
    const _Float16* src = emb + (size_t)row0 * D_FEAT;
    for (int i = lane * 8; i < TILE * D_FEAT; i += 32 * 8)
      *(v8h*)(&As[i]) = *(const v8h*)(src + i);
  }
#endif

  if (lane < TILE) {
    for (int j = 0; j < TOPK; ++j) { topv[lane][j] = -1.0e30f; topi[lane][j] = 0; }
  }

#if defined(HAVE_TDM)
#if defined(__has_builtin) && __has_builtin(__builtin_amdgcn_s_wait_tensorcnt)
  __builtin_amdgcn_s_wait_tensorcnt(0);
#else
  asm volatile("s_wait_tensorcnt 0" ::: "memory");
#endif
#endif
  __syncthreads();

  // A fragment (16x32 f16, wave32): lanes 0-15 row=lane, K {0..15};
  //                                 lanes 16-31 row=lane-16, K {16..31}
  const int arow  = lane & 15;
  const int akoff = (lane >> 4) ? 16 : 0;
  // B fragment (32x16 f16): col = lane&15, K = k0 + (lane>>4)*16 + i
  const int bkoff = (lane >> 4) << 4;
  // C/D layout: col = lane&15, row = v + (lane>>4)*8
  const int crow = (lane >> 4) * 8;
  const int ccol = lane & 15;

  for (int cg = 0; cg < NCGRP; ++cg) {
    v8f acc0 = {}, acc1 = {}, acc2 = {}, acc3 = {};
    const _Float16* b0 =
        emb + (size_t)(cg * CBLK + ccol) * D_FEAT + bkoff;
    const _Float16* b1 = b0 + (size_t)16 * D_FEAT;
    const _Float16* b2 = b0 + (size_t)32 * D_FEAT;
    const _Float16* b3 = b0 + (size_t)48 * D_FEAT;

    if (cg + 1 < NCGRP) {  // prefetch next strip's B rows
      __builtin_prefetch(b0 + (size_t)CBLK * D_FEAT, 0, 1);
      __builtin_prefetch(b2 + (size_t)CBLK * D_FEAT, 0, 1);
    }

#pragma unroll 4
    for (int k0 = 0; k0 < D_FEAT; k0 += 32) {
      const _Float16* ap = &As[arow * D_FEAT + k0 + akoff];
      v8h alo = *(const v8h*)(ap);
      v8h ahi = *(const v8h*)(ap + 8);
      v16h a = __builtin_shufflevector(alo, ahi, 0, 1, 2, 3, 4, 5, 6, 7,
                                       8, 9, 10, 11, 12, 13, 14, 15);
      acc0 = __builtin_amdgcn_wmma_f32_16x16x32_f16(
          false, a, false, *(const v16h*)(b0 + k0), (short)0, acc0, false, false);
      acc1 = __builtin_amdgcn_wmma_f32_16x16x32_f16(
          false, a, false, *(const v16h*)(b1 + k0), (short)0, acc1, false, false);
      acc2 = __builtin_amdgcn_wmma_f32_16x16x32_f16(
          false, a, false, *(const v16h*)(b2 + k0), (short)0, acc2, false, false);
      acc3 = __builtin_amdgcn_wmma_f32_16x16x32_f16(
          false, a, false, *(const v16h*)(b3 + k0), (short)0, acc3, false, false);
    }

    // Spill the 16x64 strip to LDS for the selection pass.
#pragma unroll
    for (int v = 0; v < 8; ++v) {
      tilebuf[crow + v][0 * TILE + ccol] = acc0[v];
      tilebuf[crow + v][1 * TILE + ccol] = acc1[v];
      tilebuf[crow + v][2 * TILE + ccol] = acc2[v];
      tilebuf[crow + v][3 * TILE + ccol] = acc3[v];
    }
    __syncthreads();

    // Lanes 0-15: streaming top-20 (replace current minimum).
    if (lane < TILE) {
      for (int j = 0; j < CBLK; ++j) {
        const float val = tilebuf[lane][j];
        int mi = 0;
        float mv = topv[lane][0];
#pragma unroll
        for (int t = 1; t < TOPK; ++t) {
          if (topv[lane][t] < mv) { mv = topv[lane][t]; mi = t; }
        }
        if (val > mv) { topv[lane][mi] = val; topi[lane][mi] = cg * CBLK + j; }
      }
    }
    __syncthreads();
  }

  // Symmetric thresholded scatter into the item-item block.
  if (lane < TILE) {
    const float bthr = *bptr;
    const size_t gr = (size_t)(N_USERS + row0 + lane);
    for (int t = 0; t < TOPK; ++t) {
      const float v = topv[lane][t];
      if (v > bthr) {
        const size_t gc = (size_t)(N_USERS + topi[lane][t]);
        atomicAdd(adj + gr * N_NODES + gc, v);
        atomicAdd(adj + gc * N_NODES + gr, v);
      }
    }
  }
}

// ---------------------------------------------------------------------------
extern "C" void kernel_launch(void* const* d_in, const int* in_sizes, int n_in,
                              void* d_out, int out_size, void* d_ws, size_t ws_size,
                              hipStream_t stream) {
  const float*     x    = (const float*)d_in[0];      // item_features [6144,512]
  const float*     w0   = (const float*)d_in[1];      // [512]
  const float*     w1   = (const float*)d_in[2];      // [512]
  const long long* gi   = (const long long*)d_in[3];  // graph_indices [2, NNZ] i64
  // d_in[4] graph_values: replaced by ones in reference -> unused
  // d_in[5] k: fixed 20 (TOPK)
  const float*     bptr = (const float*)d_in[6];      // threshold b

  float*     adj = (float*)d_out;                     // [12288, 12288]
  _Float16*  emb = (_Float16*)d_ws;                   // 6144*512 f16 = 6.3 MB

  // Zero the dense adjacency (604 MB -> the HBM-bound floor of this problem).
  hipMemsetAsync(d_out, 0, (size_t)out_size * sizeof(float), stream);

  embed_normalize_kernel<<<M_ITEMS, 256, 0, stream>>>(x, w0, w1, emb);
  edge_scatter_kernel<<<(NNZ + 255) / 256, 256, 0, stream>>>(gi, adj);
  simtopk_scatter_kernel<<<M_ITEMS / TILE, 32, 0, stream>>>(emb, bptr, adj);
}
